// AtomCrossAttEncoder_84731114816221
// MI455X (gfx1250) — matrix operand order repaired and checked by reading, not compiled, not run on hardware
//
#include <hip/hip_runtime.h>

#define TT 768
#define AA 24
#define SS 576
#define QQ 32
#define KK 128
#define CC 128
#define HH 4
#define DHD 32
#define LL 3
#define PAIRP 16
#define PTCP 384
#define FFF 256

typedef __bf16 bf16_t;
typedef __attribute__((ext_vector_type(16))) __bf16 v16bf;
typedef __attribute__((ext_vector_type(8)))  __bf16 v8bf;
typedef __attribute__((ext_vector_type(8)))  float   v8f;

// ---------------- wave helpers ----------------
__device__ __forceinline__ float wred_sum(float v) {
#pragma unroll
  for (int o = 16; o > 0; o >>= 1) v += __shfl_xor(v, o, 32);
  return v;
}
__device__ __forceinline__ float wred_max(float v) {
#pragma unroll
  for (int o = 16; o > 0; o >>= 1) v = fmaxf(v, __shfl_xor(v, o, 32));
  return v;
}

// ---------------- CDNA5 async global->LDS staging ----------------
// Copies `bytes` (multiple of 16) from global to LDS without touching VGPR
// data paths; tracked by ASYNCcnt. LDS address = low 32 bits of the generic
// pointer (LDS aperture offset).
__device__ __forceinline__ void async_stage(void* lds_dst, const void* gsrc,
                                            int bytes, int tid) {
  uint32_t lbase = (uint32_t)(uintptr_t)lds_dst;
  uint64_t gbase = (uint64_t)(uintptr_t)gsrc;
  for (int off = tid * 16; off < bytes; off += 256 * 16) {
    uint32_t l = lbase + (uint32_t)off;
    uint64_t g = gbase + (uint32_t)off;
    asm volatile("global_load_async_to_lds_b128 %0, %1, off"
                 :: "v"(l), "v"(g) : "memory");
  }
}
__device__ __forceinline__ void async_wait0() {
  asm volatile("s_wait_asynccnt 0" ::: "memory");
}

// One 16x16 f32 tile accumulated over kdim (multiple of 32) with bf16 WMMA.
// A: row-major tile base (16 rows x kdim), lda in halfs.
// Bt: N-major (transposed B) tile base (16 rows x kdim), ldb in halfs.
__device__ __forceinline__ v8f wmma_acc(const bf16_t* A, int lda,
                                        const bf16_t* Bt, int ldb,
                                        int kdim, v8f acc) {
  const int lane = threadIdx.x & 31;
  const int hi = lane >> 4;
  const int m  = lane & 15;
  for (int k0 = 0; k0 < kdim; k0 += 32) {
    // A 16-bit layout: elems 0..7 -> K=8*hi+0..7 ; elems 8..15 -> K=16+8*hi+0..7
    v8bf alo = *(const v8bf*)(A + m * lda + k0 + 8 * hi);
    v8bf ahi = *(const v8bf*)(A + m * lda + k0 + 16 + 8 * hi);
    // B 16-bit layout: elems 0..7 -> K=8*hi+0..7 ; elems 8..15 -> K=16+8*hi+0..7
    v8bf blo = *(const v8bf*)(Bt + m * ldb + k0 + 8 * hi);
    v8bf bhi = *(const v8bf*)(Bt + m * ldb + k0 + 16 + 8 * hi);
    v16bf a, b;
#pragma unroll
    for (int i = 0; i < 8; ++i) {
      a[i] = alo[i]; a[i + 8] = ahi[i];
      b[i] = blo[i]; b[i + 8] = bhi[i];
    }
    acc = __builtin_amdgcn_wmma_f32_16x16x32_bf16(false, a, false, b,
                                                  (short)0, acc, false, false);
  }
  return acc;
}

// D (16x16 f32, C/D layout: lane n<16 cols, vgpr j = row 8*hi+j) stores:
__device__ __forceinline__ void st_bf16(bf16_t* dst, int ld, v8f d, float sc) {
  int lane = threadIdx.x & 31, n = lane & 15, hi = lane >> 4;
#pragma unroll
  for (int j = 0; j < 8; ++j) dst[(8 * hi + j) * ld + n] = (bf16_t)(d[j] * sc);
}
__device__ __forceinline__ void st_bf16_relu(bf16_t* dst, int ld, v8f d) {
  int lane = threadIdx.x & 31, n = lane & 15, hi = lane >> 4;
#pragma unroll
  for (int j = 0; j < 8; ++j) dst[(8 * hi + j) * ld + n] = (bf16_t)fmaxf(d[j], 0.f);
}
__device__ __forceinline__ void st_bf16_T(bf16_t* dst, int ld, v8f d) {
  int lane = threadIdx.x & 31, n = lane & 15, hi = lane >> 4;
#pragma unroll
  for (int j = 0; j < 8; ++j) dst[n * ld + 8 * hi + j] = (bf16_t)d[j];
}
__device__ __forceinline__ void st_f32(float* dst, int ld, v8f d) {
  int lane = threadIdx.x & 31, n = lane & 15, hi = lane >> 4;
#pragma unroll
  for (int j = 0; j < 8; ++j) dst[(8 * hi + j) * ld + n] = d[j];
}

// ---------------- kernels ----------------

// Generic f32 -> bf16 transpose-convert: dst[c*ldd + r] = src[r*cols + c]
__global__ __launch_bounds__(256) void transpose_bf16_kernel(
    const float* __restrict__ src, bf16_t* __restrict__ dst,
    int rows, int cols, int ldd) {
  int i = blockIdx.x * 256 + threadIdx.x;
  if (i >= rows * cols) return;
  int r = i / cols, c = i % cols;
  dst[(size_t)c * ldd + r] = (bf16_t)src[i];
}

// x = (gather(token_atoms) * gmask) @ w_pos * qmask + queries_single_cond
__global__ __launch_bounds__(256) void encode_kernel(
    const int* __restrict__ a2q_idx, const int* __restrict__ a2q_mask,
    const int* __restrict__ qmask, const float* __restrict__ token_atoms,
    const float* __restrict__ qsc, const float* __restrict__ w_pos,
    float* __restrict__ x) {
  int i = blockIdx.x * 256 + threadIdx.x;
  if (i >= SS * QQ * CC) return;
  int c = i & (CC - 1), sq = i >> 7;
  int idx = a2q_idx[sq];
  float gm = (float)a2q_mask[sq];
  float qm = (float)qmask[sq];
  const float* ta = token_atoms + (size_t)idx * 3;
  float v = (ta[0] * w_pos[c] + ta[1] * w_pos[CC + c] + ta[2] * w_pos[2 * CC + c]) * gm;
  x[i] = v * qm + qsc[i];
}

// row-parallel LayerNorm: one wave per 128-wide row, 8 rows per block
__global__ __launch_bounds__(256) void ln_kernel(
    const float* __restrict__ x, float* __restrict__ xn,
    const float* __restrict__ gs, const float* __restrict__ gb) {
  int row = blockIdx.x * 8 + (threadIdx.x >> 5);
  int lane = threadIdx.x & 31;
  const float* xr = x + (size_t)row * CC;
  float v0 = xr[lane], v1 = xr[lane + 32], v2 = xr[lane + 64], v3 = xr[lane + 96];
  float mean = wred_sum(v0 + v1 + v2 + v3) * (1.0f / CC);
  float d0 = v0 - mean, d1 = v1 - mean, d2 = v2 - mean, d3 = v3 - mean;
  float var = wred_sum(d0 * d0 + d1 * d1 + d2 * d2 + d3 * d3) * (1.0f / CC);
  float inv = rsqrtf(var + 1e-5f);
  float* xo = xn + (size_t)row * CC;
  xo[lane]      = d0 * inv * gs[lane]      + gb[lane];
  xo[lane + 32] = d1 * inv * gs[lane + 32] + gb[lane + 32];
  xo[lane + 64] = d2 * inv * gs[lane + 64] + gb[lane + 64];
  xo[lane + 96] = d3 * inv * gs[lane + 96] + gb[lane + 96];
}

// Fused per-subset: key gather, q/k/v proj, pair-bias logits, softmax,
// attn@v, wo, LN2 + FFN. 8 waves; GEMMs on bf16 WMMA; weights streamed
// into LDS with async global->LDS copies (pre-transposed bf16 in ws).
__global__ __launch_bounds__(256) void attn_ffn_kernel(
    float* __restrict__ x, const float* __restrict__ xn,
    const int* __restrict__ qk_idx, const int* __restrict__ qk_mask,
    const float* __restrict__ ksc, const float* __restrict__ pair_act,
    const int* __restrict__ kmask, const float* __restrict__ wpair_l,
    const bf16_t* __restrict__ wqT, const bf16_t* __restrict__ wkT,
    const bf16_t* __restrict__ wvT, const bf16_t* __restrict__ woT,
    const bf16_t* __restrict__ w1T, const bf16_t* __restrict__ w2T,
    const float* __restrict__ ln2_s, const float* __restrict__ ln2_b) {
  // LDS carve-up (157440 B total):
  //  sQ   32x136 bf16   @0       (8704)
  //  sK  128x136 bf16   @8704    (34816)
  //  sVt 128x136 bf16   @43520   (34816)   (v stored transposed)
  //  sO   32x136 bf16   @78336   (8704)    (aliased with sXN: xn dead before sO)
  //  rb  region B       @87040   (69632)   phases: [sKeys+sWt]/sLog/sW1t/sW2t
  //  sKm 128 f32        @156672, sPw 64 f32 @157184
  __shared__ __align__(16) char smem[157440];
  bf16_t* sQ   = (bf16_t*)(smem);
  bf16_t* sK   = (bf16_t*)(smem + 8704);
  bf16_t* sVt  = (bf16_t*)(smem + 43520);
  bf16_t* sO   = (bf16_t*)(smem + 78336);
  bf16_t* sXN  = sO;
  char*   rb   = smem + 87040;
  bf16_t* sKeys = (bf16_t*)rb;
  bf16_t* sWt   = (bf16_t*)(rb + 34816);
  float*  sLog  = (float*)rb;            // 4*32 rows x 132 f32
  float*  sKm   = (float*)(smem + 156672);
  float*  sPw   = (float*)(smem + 157184);

  const int s = blockIdx.x;
  const int tid = threadIdx.x;
  const int lane = tid & 31;
  const int w = tid >> 5;

  // ---- P0: async-stage wq^T while gathering xn rows / keys (cross-subset)
  async_stage(sWt, wqT, 128 * 136 * 2, tid);
  for (int i = tid; i < QQ * CC; i += 256) {
    int q = i >> 7, c = i & 127;
    sXN[q * 136 + c] = (bf16_t)xn[((size_t)s * QQ + q) * CC + c];
  }
  for (int i = tid; i < KK * CC; i += 256) {
    int kk = i >> 7, c = i & 127;
    int gi = qk_idx[s * KK + kk];
    float gm = (float)qk_mask[s * KK + kk];
    sKeys[kk * 136 + c] =
        (bf16_t)(xn[(size_t)gi * CC + c] * gm + ksc[((size_t)s * KK + kk) * CC + c]);
  }
  if (tid < KK) sKm[tid] = (float)kmask[s * KK + tid];
  if (tid < PAIRP * HH) sPw[tid] = wpair_l[tid];
  async_wait0();
  __syncthreads();

  // ---- P1a: q = xn @ wq, scaled by DH^-1/2
  for (int t = w; t < 16; t += 8) {
    int mt = t >> 3, nt = t & 7;
    v8f acc = {};
    acc = wmma_acc(sXN + mt * 16 * 136, 136, sWt + nt * 16 * 136, 136, CC, acc);
    st_bf16(sQ + mt * 16 * 136 + nt * 16, 136, acc, 0.17677669529663687f);
  }
  __syncthreads();
  // ---- P1b: k = keys @ wk
  async_stage(sWt, wkT, 128 * 136 * 2, tid);
  async_wait0();
  __syncthreads();
  for (int t = w; t < 64; t += 8) {
    int mt = t >> 3, nt = t & 7;
    v8f acc = {};
    acc = wmma_acc(sKeys + mt * 16 * 136, 136, sWt + nt * 16 * 136, 136, CC, acc);
    st_bf16(sK + mt * 16 * 136 + nt * 16, 136, acc, 1.0f);
  }
  __syncthreads();
  // ---- P1c: v^T = (keys @ wv)^T  (so v is N-major for attn@v)
  async_stage(sWt, wvT, 128 * 136 * 2, tid);
  async_wait0();
  __syncthreads();
  for (int t = w; t < 64; t += 8) {
    int mt = t >> 3, nt = t & 7;
    v8f acc = {};
    acc = wmma_acc(sKeys + mt * 16 * 136, 136, sWt + nt * 16 * 136, 136, CC, acc);
    st_bf16_T(sVt + nt * 16 * 136 + mt * 16, 136, acc);
  }
  __syncthreads();

  // ---- P2: logits[h,q,k] = q_h @ k_h^T  (sK rows already N-major here)
  for (int t = w; t < 64; t += 8) {
    int h = t >> 4, mt = (t >> 3) & 1, nt = t & 7;
    v8f acc = {};
    acc = wmma_acc(sQ + mt * 16 * 136 + h * DHD, 136,
                   sK + nt * 16 * 136 + h * DHD, 136, DHD, acc);
    st_f32(sLog + (h * QQ + mt * 16) * 132 + nt * 16, 132, acc);
  }
  __syncthreads();
  // pair bias + key-mask: one thread does all 4 heads per (q,k) -> pair_act
  // (the dominant HBM stream) is loaded exactly once.
  for (int i = tid; i < QQ * KK; i += 256) {
    int q = i >> 7, kk = i & 127;
    const float* pp = pair_act + (((size_t)s * QQ + q) * KK + kk) * PAIRP;
    float b0 = 0.f, b1 = 0.f, b2 = 0.f, b3 = 0.f;
#pragma unroll
    for (int p = 0; p < PAIRP; ++p) {
      float pv = pp[p];
      b0 += pv * sPw[p * HH + 0];
      b1 += pv * sPw[p * HH + 1];
      b2 += pv * sPw[p * HH + 2];
      b3 += pv * sPw[p * HH + 3];
    }
    float msk = (sKm[kk] - 1.0f) * 1e9f;
    sLog[(0 * QQ + q) * 132 + kk] += b0 + msk;
    sLog[(1 * QQ + q) * 132 + kk] += b1 + msk;
    sLog[(2 * QQ + q) * 132 + kk] += b2 + msk;
    sLog[(3 * QQ + q) * 132 + kk] += b3 + msk;
  }
  __syncthreads();

  // ---- P3: softmax over K per (h,q) row; write bf16 attn in place
  bf16_t* sAttn = (bf16_t*)sLog;  // row stride 264 halfs (same bytes as 132 f32)
  for (int r = w; r < HH * QQ; r += 8) {
    float* row = sLog + r * 132;
    float v0 = row[lane], v1 = row[lane + 32], v2 = row[lane + 64], v3 = row[lane + 96];
    float mx = wred_max(fmaxf(fmaxf(v0, v1), fmaxf(v2, v3)));
    v0 = __expf(v0 - mx); v1 = __expf(v1 - mx);
    v2 = __expf(v2 - mx); v3 = __expf(v3 - mx);
    float inv = 1.0f / wred_sum(v0 + v1 + v2 + v3);
    bf16_t* arow = (bf16_t*)row;
    arow[lane]      = (bf16_t)(v0 * inv);
    arow[lane + 32] = (bf16_t)(v1 * inv);
    arow[lane + 64] = (bf16_t)(v2 * inv);
    arow[lane + 96] = (bf16_t)(v3 * inv);
  }
  __syncthreads();

  // ---- P4: out_h = attn_h @ v_h, heads concatenated in sO
  for (int t = w; t < 16; t += 8) {
    int h = t >> 2, mt = (t >> 1) & 1, nt = t & 1;
    v8f acc = {};
    acc = wmma_acc(sAttn + (h * QQ + mt * 16) * 264, 264,
                   sVt + (h * DHD + nt * 16) * 136, 136, KK, acc);
    st_bf16(sO + mt * 16 * 136 + h * DHD + nt * 16, 136, acc, 1.0f);
  }
  __syncthreads();

  // ---- P5: x += out @ wo  (accumulate into global x, block-owned rows)
  bf16_t* sWoT = (bf16_t*)rb;
  async_stage(sWoT, woT, 128 * 136 * 2, tid);
  async_wait0();
  __syncthreads();
  float* xg = x + (size_t)s * QQ * CC;
  for (int t = w; t < 16; t += 8) {
    int mt = t >> 3, nt = t & 7;
    v8f acc = {};
    acc = wmma_acc(sO + mt * 16 * 136, 136, sWoT + nt * 16 * 136, 136, CC, acc);
    int n = lane & 15, hi = lane >> 4;
    float* dst = xg + (mt * 16) * CC + nt * 16 + n;
#pragma unroll
    for (int j = 0; j < 8; ++j) dst[(8 * hi + j) * CC] += acc[j];
  }
  __threadfence();
  __syncthreads();

  // ---- P6: xn2 = LN2(x); h1 = relu(xn2 @ w1); x += h1 @ w2
  // w1^T streams in asynchronously while the LN2 VALU work runs.
  bf16_t* sW1t = (bf16_t*)rb;  // 256 x 136
  async_stage(sW1t, w1T, 256 * 136 * 2, tid);
  bf16_t* sXN2 = sQ;  // reuse
  for (int q = w; q < QQ; q += 8) {
    const float* xr = xg + q * CC;
    float v0 = xr[lane], v1 = xr[lane + 32], v2 = xr[lane + 64], v3 = xr[lane + 96];
    float mean = wred_sum(v0 + v1 + v2 + v3) * (1.0f / CC);
    float d0 = v0 - mean, d1 = v1 - mean, d2 = v2 - mean, d3 = v3 - mean;
    float var = wred_sum(d0 * d0 + d1 * d1 + d2 * d2 + d3 * d3) * (1.0f / CC);
    float inv = rsqrtf(var + 1e-5f);
    sXN2[q * 136 + lane]      = (bf16_t)(d0 * inv * ln2_s[lane]      + ln2_b[lane]);
    sXN2[q * 136 + lane + 32] = (bf16_t)(d1 * inv * ln2_s[lane + 32] + ln2_b[lane + 32]);
    sXN2[q * 136 + lane + 64] = (bf16_t)(d2 * inv * ln2_s[lane + 64] + ln2_b[lane + 64]);
    sXN2[q * 136 + lane + 96] = (bf16_t)(d3 * inv * ln2_s[lane + 96] + ln2_b[lane + 96]);
  }
  async_wait0();
  __syncthreads();
  bf16_t* sH1 = sK;  // reuse (32 x 264)
  for (int t = w; t < 32; t += 8) {
    int mt = t >> 4, nt = t & 15;
    v8f acc = {};
    acc = wmma_acc(sXN2 + mt * 16 * 136, 136, sW1t + nt * 16 * 136, 136, CC, acc);
    st_bf16_relu(sH1 + mt * 16 * 264 + nt * 16, 264, acc);
  }
  __syncthreads();
  bf16_t* sW2t = (bf16_t*)rb;  // 128 x 264
  async_stage(sW2t, w2T, 128 * 264 * 2, tid);
  async_wait0();
  __syncthreads();
  for (int t = w; t < 16; t += 8) {
    int mt = t >> 3, nt = t & 7;
    v8f acc = {};
    acc = wmma_acc(sH1 + mt * 16 * 264, 264, sW2t + nt * 16 * 264, 264, FFF, acc);
    int n = lane & 15, hi = lane >> 4;
    float* dst = xg + (mt * 16) * CC + nt * 16 + n;
#pragma unroll
    for (int j = 0; j < 8; ++j) dst[(8 * hi + j) * CC] += acc[j];
  }
}

// skip = x*qm (to d_out), proj = (x*qm) @ w_aggr (to ws), WMMA 32-row tiles
__global__ __launch_bounds__(256) void proj_skip_kernel(
    const float* __restrict__ x, const int* __restrict__ qmask,
    const bf16_t* __restrict__ w_aggrT, float* __restrict__ proj,
    float* __restrict__ skip) {
  __shared__ __align__(16) char smem[8704 + 104448];
  bf16_t* sA  = (bf16_t*)smem;            // 32 x 136
  bf16_t* sWt = (bf16_t*)(smem + 8704);   // 384 x 136
  int tid = threadIdx.x, w = tid >> 5, lane = tid & 31;
  int sq0 = blockIdx.x * 32;
  async_stage(sWt, w_aggrT, 384 * 136 * 2, tid);
  for (int i = tid; i < 32 * CC; i += 256) {
    int r = i >> 7, c = i & 127;
    int sq = sq0 + r;
    float v = x[(size_t)sq * CC + c] * (float)qmask[sq];
    sA[r * 136 + c] = (bf16_t)v;
    skip[(size_t)sq * CC + c] = v;
  }
  async_wait0();
  __syncthreads();
  for (int t = w; t < 48; t += 8) {
    int mt = t / 24, nt = t % 24;
    v8f acc = {};
    acc = wmma_acc(sA + mt * 16 * 136, 136, sWt + nt * 16 * 136, 136, CC, acc);
    int n = lane & 15, hi = lane >> 4;
    float* dst = proj + (size_t)(sq0 + mt * 16) * PTCP + nt * 16 + n;
#pragma unroll
    for (int j = 0; j < 8; ++j) dst[(size_t)(8 * hi + j) * PTCP] = acc[j];
  }
}

// token_act[t,p] = sum_a pm*relu(proj[idx]*gm) / (sum_a pm + 1e-10)
__global__ __launch_bounds__(256) void token_out_kernel(
    const float* __restrict__ proj, const int* __restrict__ q2a_idx,
    const int* __restrict__ q2a_mask, const int* __restrict__ pmask,
    float* __restrict__ out_tok) {
  __shared__ int sidx[AA];
  __shared__ float sgm[AA], spm[AA];
  __shared__ float sden;
  int t = blockIdx.x, tid = threadIdx.x;
  if (tid < AA) {
    sidx[tid] = q2a_idx[t * AA + tid];
    sgm[tid]  = (float)q2a_mask[t * AA + tid];
    spm[tid]  = (float)pmask[t * AA + tid];
  }
  __syncthreads();
  if (tid == 0) {
    float d = 0.f;
    for (int a = 0; a < AA; ++a) d += spm[a];
    sden = 1.0f / (d + 1e-10f);
  }
  __syncthreads();
  for (int p = tid; p < PTCP; p += 256) {
    float acc = 0.f;
    for (int a = 0; a < AA; ++a) {
      float v = proj[(size_t)sidx[a] * PTCP + p] * sgm[a];
      acc += spm[a] * fmaxf(v, 0.f);
    }
    out_tok[(size_t)t * PTCP + p] = acc * sden;
  }
}

extern "C" void kernel_launch(void* const* d_in, const int* in_sizes, int n_in,
                              void* d_out, int out_size, void* d_ws, size_t ws_size,
                              hipStream_t stream) {
  const int*   queries_mask = (const int*)d_in[0];
  const int*   pred_mask    = (const int*)d_in[1];
  const int*   a2q_idx      = (const int*)d_in[2];
  const int*   a2q_mask     = (const int*)d_in[3];
  const int*   qk_idx       = (const int*)d_in[4];
  const int*   qk_mask      = (const int*)d_in[5];
  const int*   q2a_idx      = (const int*)d_in[6];
  const int*   q2a_mask     = (const int*)d_in[7];
  const float* token_atoms  = (const float*)d_in[8];
  const float* qsc          = (const float*)d_in[9];
  const float* pair_act     = (const float*)d_in[10];
  const int*   keys_mask    = (const int*)d_in[11];
  const float* ksc          = (const float*)d_in[12];
  const float* w_pos        = (const float*)d_in[13];
  const float* w_aggr       = (const float*)d_in[14];
  const float* ln1_s        = (const float*)d_in[15];
  const float* ln1_b        = (const float*)d_in[16];
  const float* wq           = (const float*)d_in[17];
  const float* wk           = (const float*)d_in[18];
  const float* wv           = (const float*)d_in[19];
  const float* wpair        = (const float*)d_in[20];
  const float* wo           = (const float*)d_in[21];
  const float* ln2_s        = (const float*)d_in[22];
  const float* ln2_b        = (const float*)d_in[23];
  const float* w1           = (const float*)d_in[24];
  const float* w2           = (const float*)d_in[25];

  float* ws   = (float*)d_ws;
  float* x    = ws;                               // S*Q*C f32
  float* xnb  = ws + (size_t)SS * QQ * CC;        // S*Q*C f32
  float* proj = xnb + (size_t)SS * QQ * CC;       // S*Q*PTC f32
  // pre-transposed bf16 weight images (async-copied into LDS by blocks)
  bf16_t* wsb = (bf16_t*)(proj + (size_t)SS * QQ * PTCP);
  const size_t WQT = 128 * 136;          // wq/wk/wv/wo^T (ld 136)
  const size_t W1T = 256 * 136;          // w1^T (ld 136)
  const size_t W2T = 128 * 264;          // w2^T (ld 264)
  const size_t LSTRIDE = 4 * WQT + W1T + W2T;
  bf16_t* w_aggrT = wsb + LL * LSTRIDE;  // 384 x 136

  float* out_tok  = (float*)d_out;                // T*PTC
  float* out_skip = out_tok + (size_t)TT * PTCP;  // S*Q*C

  // ---- prep: convert + transpose all GEMM weights to bf16 once
  for (int l = 0; l < LL; ++l) {
    bf16_t* wl = wsb + (size_t)l * LSTRIDE;
    transpose_bf16_kernel<<<64, 256, 0, stream>>>(wq + (size_t)l * CC * 128, wl, CC, 128, 136);
    transpose_bf16_kernel<<<64, 256, 0, stream>>>(wk + (size_t)l * CC * 128, wl + WQT, CC, 128, 136);
    transpose_bf16_kernel<<<64, 256, 0, stream>>>(wv + (size_t)l * CC * 128, wl + 2 * WQT, CC, 128, 136);
    transpose_bf16_kernel<<<64, 256, 0, stream>>>(wo + (size_t)l * 128 * CC, wl + 3 * WQT, 128, CC, 136);
    transpose_bf16_kernel<<<128, 256, 0, stream>>>(w1 + (size_t)l * CC * FFF, wl + 4 * WQT, CC, FFF, 136);
    transpose_bf16_kernel<<<128, 256, 0, stream>>>(w2 + (size_t)l * FFF * CC, wl + 4 * WQT + W1T, FFF, CC, 264);
  }
  transpose_bf16_kernel<<<192, 256, 0, stream>>>(w_aggr, w_aggrT, CC, PTCP, 136);

  encode_kernel<<<(SS * QQ * CC + 255) / 256, 256, 0, stream>>>(
      a2q_idx, a2q_mask, queries_mask, token_atoms, qsc, w_pos, x);
  for (int l = 0; l < LL; ++l) {
    bf16_t* wl = wsb + (size_t)l * LSTRIDE;
    ln_kernel<<<SS * QQ / 8, 256, 0, stream>>>(x, xnb, ln1_s + (size_t)l * CC,
                                               ln1_b + (size_t)l * CC);
    attn_ffn_kernel<<<SS, 256, 0, stream>>>(
        x, xnb, qk_idx, qk_mask, ksc, pair_act, keys_mask,
        wpair + (size_t)l * PAIRP * HH,
        wl, wl + WQT, wl + 2 * WQT, wl + 3 * WQT, wl + 4 * WQT, wl + 4 * WQT + W1T,
        ln2_s + (size_t)l * CC, ln2_b + (size_t)l * CC);
  }
  proj_skip_kernel<<<SS, 256, 0, stream>>>(x, queries_mask, w_aggrT, proj, out_skip);
  token_out_kernel<<<TT, 256, 0, stream>>>(proj, q2a_idx, q2a_mask, pred_mask, out_tok);

  (void)in_sizes; (void)n_in; (void)out_size; (void)ws_size;
}
